// VQModel_56573309223736
// MI455X (gfx1250) — matrix operand order repaired and checked by reading, not compiled, not run on hardware
//
#include <hip/hip_runtime.h>
#include <hip/hip_bf16.h>

typedef __bf16 bf16_t;
typedef __attribute__((ext_vector_type(16))) __bf16 v16bf;
typedef __attribute__((ext_vector_type(8)))  float  v8f;

#define BM 128
#define BN 128
#define BK 32
#define LDA 40   // padded LDS row stride (bf16 elems): 16 distinct bank groups

union FragBF { uint4 u[2]; v16bf v; };

// ---------------------------------------------------------------------------
// Shared epilogue: mode 0 -> bias/relu store (+ optional bf16 mirror)
//                  mode 1 -> packed (dist,idx) argmin via atomicMin_u64
// ---------------------------------------------------------------------------
__device__ __forceinline__ void gemm_epilogue(
    v8f (&acc)[2][4], int m0, int n0, int wm, int wn, int lane,
    int Cout, int dstStride, const float* __restrict__ bias,
    float alpha, int relu, int mode,
    float* __restrict__ dst, bf16_t* __restrict__ dstb,
    unsigned long long* __restrict__ minbuf)
{
    if (mode == 0) {
#pragma unroll
        for (int nt = 0; nt < 4; ++nt) {
            int n = n0 + wn * 64 + nt * 16 + (lane & 15);
            if (n < Cout) {
                float b = bias[n];
#pragma unroll
                for (int mt = 0; mt < 2; ++mt)
#pragma unroll
                    for (int g = 0; g < 8; ++g) {
                        int m = m0 + wm * 32 + mt * 16 + ((lane >> 4) << 3) + g;
                        float v = acc[mt][nt][g] * alpha + b;
                        if (relu) v = fmaxf(v, 0.f);
                        dst[(size_t)m * dstStride + n] = v;
                        if (dstb) dstb[(size_t)m * dstStride + n] = (bf16_t)v;
                    }
            }
        }
    } else {
#pragma unroll
        for (int mt = 0; mt < 2; ++mt)
#pragma unroll
            for (int g = 0; g < 8; ++g) {
                unsigned long long best = ~0ull;
#pragma unroll
                for (int nt = 0; nt < 4; ++nt) {
                    int n = n0 + wn * 64 + nt * 16 + (lane & 15);
                    float v = acc[mt][nt][g] * alpha + bias[n];
                    unsigned u = __float_as_uint(v);
                    u = (u & 0x80000000u) ? ~u : (u | 0x80000000u);
                    unsigned long long p = ((unsigned long long)u << 32) | (unsigned)n;
                    best = (p < best) ? p : best;
                }
#pragma unroll
                for (int s = 1; s < 16; s <<= 1) {
                    unsigned lo = __shfl_xor((unsigned)best, s, 16);
                    unsigned hi = __shfl_xor((unsigned)(best >> 32), s, 16);
                    unsigned long long o = ((unsigned long long)hi << 32) | lo;
                    best = (o < best) ? o : best;
                }
                if ((lane & 15) == 0) {
                    int m = m0 + wm * 32 + mt * 16 + ((lane >> 4) << 3) + g;
                    atomicMin(&minbuf[m], best);
                }
            }
    }
}

// ---------------------------------------------------------------------------
// Generic implicit-GEMM conv / convT kernel (gathered im2col A), bf16 WMMA.
// ---------------------------------------------------------------------------
__global__ __launch_bounds__(256)
void k_gemm(const float* __restrict__ src,
            int H, int W, int Cin, int cinShift,
            int kh, int kw, int stride, int pad, int tmode,
            int Ho, int Wo,
            const bf16_t* __restrict__ Bw, int Kpad, int Cout, int dstStride,
            const float* __restrict__ bias, float alpha, int relu, int mode,
            float* __restrict__ dst, bf16_t* __restrict__ dstb,
            unsigned long long* __restrict__ minbuf)
{
    __shared__ bf16_t As[BM * LDA];

    const int tid  = threadIdx.x;
    const int lane = tid & 31;
    const int wid  = tid >> 5;
    const int wm   = wid & 3;
    const int wn   = wid >> 2;
    const int m0   = blockIdx.x * BM;
    const int n0   = blockIdx.y * BN;
    const int HoWo = Ho * Wo;
    const int Kreal = Cin * kh * kw;

    const int rbase = tid >> 4;
    const int jbase = (tid & 15) * 2;
    int nb[8], oyA[8], oxA[8];
#pragma unroll
    for (int i = 0; i < 8; ++i) {
        int m    = m0 + i * 16 + rbase;
        int nimg = m / HoWo;
        int r    = m - nimg * HoWo;
        oyA[i]   = r / Wo;
        oxA[i]   = r - oyA[i] * Wo;
        nb[i]    = nimg * (H * W);
    }

    v8f acc[2][4];
#pragma unroll
    for (int a = 0; a < 2; ++a)
#pragma unroll
        for (int b = 0; b < 4; ++b)
            acc[a][b] = (v8f){0.f,0.f,0.f,0.f,0.f,0.f,0.f,0.f};

    const int aoff = (lane >> 4) << 3;

    for (int kb = 0; kb < Kpad; kb += BK) {
        __syncthreads();
#pragma unroll
        for (int i = 0; i < 8; ++i) {
            int rofs = (i * 16 + rbase) * LDA + jbase;
#pragma unroll
            for (int e = 0; e < 2; ++e) {
                int kk = kb + jbase + e;
                float v = 0.f;
                if (kk < Kreal) {
                    int tap, c;
                    if (cinShift >= 0) { tap = kk >> cinShift; c = kk & (Cin - 1); }
                    else               { tap = kk / Cin;       c = kk - tap * Cin; }
                    int ky = tap / kw;
                    int kx = tap - ky * kw;
                    int iy, ix; bool ok;
                    if (!tmode) {
                        iy = oyA[i] * stride - pad + ky;
                        ix = oxA[i] * stride - pad + kx;
                        ok = ((unsigned)iy < (unsigned)H) && ((unsigned)ix < (unsigned)W);
                    } else {
                        int u = oyA[i] - 2 + ky;
                        int t = oxA[i] - 2 + kx;
                        ok = (u >= 0) && (t >= 0) && !(u & 1) && !(t & 1);
                        iy = u >> 1; ix = t >> 1;
                        ok = ok && (iy < H) && (ix < W);
                    }
                    if (ok) v = src[((size_t)nb[i] + iy * W + ix) * Cin + c];
                }
                As[rofs + e] = (bf16_t)v;
            }
        }
        __syncthreads();

        FragBF af[2], bfm[4];
#pragma unroll
        for (int mt = 0; mt < 2; ++mt) {
            const bf16_t* pa = &As[(wm * 32 + mt * 16 + (lane & 15)) * LDA + aoff];
            af[mt].u[0] = *(const uint4*)(pa);
            af[mt].u[1] = *(const uint4*)(pa + 16);
        }
#pragma unroll
        for (int nt = 0; nt < 4; ++nt) {
            int n = n0 + wn * 64 + nt * 16 + (lane & 15);
            const bf16_t* pb = Bw + (size_t)n * Kpad + kb + ((lane >> 4) << 3);
            bfm[nt].u[0] = *(const uint4*)(pb);
            bfm[nt].u[1] = *(const uint4*)(pb + 16);
            if (kb + BK < Kpad) __builtin_prefetch(pb + BK, 0, 3);
        }
#pragma unroll
        for (int mt = 0; mt < 2; ++mt)
#pragma unroll
            for (int nt = 0; nt < 4; ++nt)
                acc[mt][nt] = __builtin_amdgcn_wmma_f32_16x16x32_bf16(
                    false, af[mt].v, false, bfm[nt].v, (short)0, acc[mt][nt], false, false);
    }

    gemm_epilogue(acc, m0, n0, wm, wn, lane, Cout, dstStride, bias, alpha, relu,
                  mode, dst, dstb, minbuf);
}

// ---------------------------------------------------------------------------
// 1x1 / linear GEMM: A is contiguous bf16 [M][Kpad]. A tiles staged with
// double-buffered GLOBAL_LOAD_ASYNC_TO_LDS_B128 (ASYNCcnt), next slab in
// flight while WMMAs consume the current one.
// ---------------------------------------------------------------------------
__global__ __launch_bounds__(256)
void k_gemm_lin(const bf16_t* __restrict__ A,
                const bf16_t* __restrict__ Bw,
                int Kpad, int Cout, int dstStride,
                const float* __restrict__ bias, float alpha, int relu, int mode,
                float* __restrict__ dst, bf16_t* __restrict__ dstb,
                unsigned long long* __restrict__ minbuf)
{
    __shared__ bf16_t As[2][BM * LDA];

    const int tid  = threadIdx.x;
    const int lane = tid & 31;
    const int wid  = tid >> 5;
    const int wm   = wid & 3;
    const int wn   = wid >> 2;
    const int m0   = blockIdx.x * BM;
    const int n0   = blockIdx.y * BN;

    // 512 x 16B chunks per 128x32 bf16 tile; 2 per thread
    unsigned ldsb[2][2], gofs[2];
#pragma unroll
    for (int j = 0; j < 2; ++j) {
        int chunk = tid + j * 256;
        int row = chunk >> 2, q = chunk & 3;
        ldsb[0][j] = (unsigned)(uintptr_t)(&As[0][0]) + row * (LDA * 2) + q * 16;
        ldsb[1][j] = (unsigned)(uintptr_t)(&As[1][0]) + row * (LDA * 2) + q * 16;
        gofs[j]    = (unsigned)(((size_t)(m0 + row) * Kpad + q * 8) * 2);
    }

    v8f acc[2][4];
#pragma unroll
    for (int a = 0; a < 2; ++a)
#pragma unroll
        for (int b = 0; b < 4; ++b)
            acc[a][b] = (v8f){0.f,0.f,0.f,0.f,0.f,0.f,0.f,0.f};

    const int aoff = (lane >> 4) << 3;

    // prologue: stage K-slab 0 into buffer 0
#pragma unroll
    for (int j = 0; j < 2; ++j)
        asm volatile("global_load_async_to_lds_b128 %0, %1, %2"
                     :: "v"(ldsb[0][j]), "v"(gofs[j]), "s"(A) : "memory");

    int buf = 0;
    for (int kb = 0; kb < Kpad; kb += BK) {
        if (kb + BK < Kpad) {
#pragma unroll
            for (int j = 0; j < 2; ++j)
                asm volatile("global_load_async_to_lds_b128 %0, %1, %2"
                             :: "v"(ldsb[buf ^ 1][j]),
                                "v"(gofs[j] + (unsigned)((kb + BK) * 2)),
                                "s"(A) : "memory");
            asm volatile("s_wait_asynccnt 2" ::: "memory");   // current slab done
        } else {
            asm volatile("s_wait_asynccnt 0" ::: "memory");
        }
        __syncthreads();   // all waves' async writes for current buf visible

        FragBF af[2], bfm[4];
#pragma unroll
        for (int mt = 0; mt < 2; ++mt) {
            const bf16_t* pa = &As[buf][(wm * 32 + mt * 16 + (lane & 15)) * LDA + aoff];
            af[mt].u[0] = *(const uint4*)(pa);
            af[mt].u[1] = *(const uint4*)(pa + 16);
        }
#pragma unroll
        for (int nt = 0; nt < 4; ++nt) {
            int n = n0 + wn * 64 + nt * 16 + (lane & 15);
            const bf16_t* pb = Bw + (size_t)n * Kpad + kb + ((lane >> 4) << 3);
            bfm[nt].u[0] = *(const uint4*)(pb);
            bfm[nt].u[1] = *(const uint4*)(pb + 16);
            if (kb + BK < Kpad) __builtin_prefetch(pb + BK, 0, 3);
        }
#pragma unroll
        for (int mt = 0; mt < 2; ++mt)
#pragma unroll
            for (int nt = 0; nt < 4; ++nt)
                acc[mt][nt] = __builtin_amdgcn_wmma_f32_16x16x32_bf16(
                    false, af[mt].v, false, bfm[nt].v, (short)0, acc[mt][nt], false, false);

        __syncthreads();   // all reads of buf done before it is re-filled
        buf ^= 1;
    }

    gemm_epilogue(acc, m0, n0, wm, wn, lane, Cout, dstStride, bias, alpha, relu,
                  mode, dst, dstb, minbuf);
}

// ---------------------------------------------------------------------------
__global__ void k_wtrans(const float* __restrict__ w, bf16_t* __restrict__ B,
                         int Cout, int Cin, int kh, int kw, int Kpad, int CoutPad, int tmode)
{
    int i = blockIdx.x * 256 + threadIdx.x;
    if (i >= CoutPad * Kpad) return;
    int o = i / Kpad, kk = i - o * Kpad;
    float v = 0.f;
    if (o < Cout && kk < Cin * kh * kw) {
        int tap = kk / Cin, c = kk - tap * Cin;
        int ky = tap / kw, kx = tap - ky * kw;
        if (!tmode) v = w[((size_t)(o * Cin + c) * kh + ky) * kw + kx];
        else        v = w[((size_t)(c * Cout + o) * kh + (kh - 1 - ky)) * kw + (kw - 1 - kx)];
    }
    B[(size_t)o * Kpad + kk] = (bf16_t)v;
}

__global__ void k_rownorm(const float* __restrict__ cb, float* __restrict__ cn, int K, int E)
{
    int k = blockIdx.x * 256 + threadIdx.x;
    if (k >= K) return;
    const float* r = cb + (size_t)k * E;
    float s = 0.f;
    for (int c = 0; c < E; ++c) s += r[c] * r[c];
    cn[k] = s;
}

__global__ void k_init(unsigned long long* __restrict__ minbuf, float* __restrict__ loss, int n)
{
    int i = blockIdx.x * 256 + threadIdx.x;
    if (i < n) minbuf[i] = ~0ull;
    if (i == 0) *loss = 0.f;
}

__global__ void k_nchw_to_cl(const float* __restrict__ x, float* __restrict__ y,
                             int C, int H, int W, int total)
{
    int i = blockIdx.x * 256 + threadIdx.x;
    if (i >= total) return;
    int hw = i % (H * W);
    int t  = i / (H * W);
    int c  = t % C;
    int n  = t / C;
    y[((size_t)n * H * W + hw) * C + c] = x[i];
}

__global__ void k_cl_to_nchw(const float* __restrict__ cl, float* __restrict__ out,
                             int C, int H, int W, int total)
{
    int i = blockIdx.x * 256 + threadIdx.x;
    if (i >= total) return;
    int hw = i % (H * W);
    int t  = i / (H * W);
    int c  = t % C;
    int n  = t / C;
    out[i] = cl[((size_t)n * H * W + hw) * C + c];
}

__global__ void k_vq_finish(const unsigned long long* __restrict__ minbuf,
                            const float* __restrict__ codebook,
                            const float* __restrict__ h,
                            float* __restrict__ quant,
                            bf16_t* __restrict__ quantb,
                            float* __restrict__ idx_out,
                            float* __restrict__ lossAcc, int E)
{
    __shared__ float red[256];
    int row = blockIdx.x;
    int c   = threadIdx.x;
    unsigned idx = (unsigned)(minbuf[row] & 0xffffffffull);
    float q = codebook[(size_t)idx * E + c];
    float z = h[(size_t)row * E + c];
    quant[(size_t)row * E + c]  = q;
    quantb[(size_t)row * E + c] = (bf16_t)q;
    float d = q - z;
    red[c] = d * d;
    __syncthreads();
    for (int s = 128; s > 0; s >>= 1) {
        if (c < s) red[c] += red[c + s];
        __syncthreads();
    }
    if (c == 0) {
        atomicAdd(lossAcc, red[0]);
        idx_out[row] = (float)idx;
    }
}

__global__ void k_loss_final(const float* __restrict__ lossAcc, float* __restrict__ out, float scale)
{
    out[0] = lossAcc[0] * scale;
}

// ---------------------------------------------------------------------------
extern "C" void kernel_launch(void* const* d_in, const int* in_sizes, int n_in,
                              void* d_out, int out_size, void* d_ws, size_t ws_size,
                              hipStream_t stream)
{
    const float* x       = (const float*)d_in[0];
    const float* enc_w1  = (const float*)d_in[1];  const float* enc_b1  = (const float*)d_in[2];
    const float* enc_w2  = (const float*)d_in[3];  const float* enc_b2  = (const float*)d_in[4];
    const float* enc_w3  = (const float*)d_in[5];  const float* enc_b3  = (const float*)d_in[6];
    const float* qconv_w = (const float*)d_in[7];  const float* qconv_b = (const float*)d_in[8];
    const float* codebook= (const float*)d_in[9];
    const float* pq_w    = (const float*)d_in[10]; const float* pq_b    = (const float*)d_in[11];
    const float* dec_w1  = (const float*)d_in[12]; const float* dec_b1  = (const float*)d_in[13];
    const float* dt_w2   = (const float*)d_in[14]; const float* dt_b2   = (const float*)d_in[15];
    const float* dt_w3   = (const float*)d_in[16]; const float* dt_b3   = (const float*)d_in[17];

    float* out = (float*)d_out;
    const int NI = 4, Hc = 128, E = 256, K = 8192;
    const int RECON = 4 * 3 * 256 * 256;
    const int NROW  = 4 * 64 * 64;

    char* base = (char*)d_ws; size_t off = 0;
    auto alloc = [&](size_t bytes) -> void* {
        void* p = base + off; off = (off + bytes + 255) & ~(size_t)255; return p;
    };
    float* x_cl  = (float*)alloc((size_t)RECON * 4);            float* recon = x_cl;
    float* a1    = (float*)alloc((size_t)65536 * 128 * 4);      float* d2    = a1;
    float* a2    = (float*)alloc((size_t)16384 * 128 * 4);      float* d1    = a2;
    float* a3    = (float*)alloc((size_t)16384 * 128 * 4);      float* qb    = a3;
    float* h     = (float*)alloc((size_t)NROW * E * 4);
    float* quant = (float*)alloc((size_t)NROW * E * 4);
    bf16_t* a3b  = (bf16_t*)alloc((size_t)NROW * 128 * 2);
    bf16_t* hb   = (bf16_t*)alloc((size_t)NROW * E * 2);
    bf16_t* qtb  = (bf16_t*)alloc((size_t)NROW * E * 2);
    unsigned long long* minbuf = (unsigned long long*)alloc((size_t)NROW * 8);
    float* cnorm = (float*)alloc((size_t)K * 4);
    float* lossA = (float*)alloc(4);
    bf16_t* Benc1 = (bf16_t*)alloc((size_t)128 * 64   * 2);
    bf16_t* Benc2 = (bf16_t*)alloc((size_t)128 * 2048 * 2);
    bf16_t* Benc3 = (bf16_t*)alloc((size_t)128 * 1152 * 2);
    bf16_t* Bqc   = (bf16_t*)alloc((size_t)256 * 128  * 2);
    bf16_t* Bcb   = (bf16_t*)alloc((size_t)8192 * 256 * 2);
    bf16_t* Bpq   = (bf16_t*)alloc((size_t)128 * 256  * 2);
    bf16_t* Bdc1  = (bf16_t*)alloc((size_t)128 * 1152 * 2);
    bf16_t* Bdt2  = (bf16_t*)alloc((size_t)128 * 2048 * 2);
    bf16_t* Bdt3  = (bf16_t*)alloc((size_t)128 * 2048 * 2);

    dim3 b256(256);

    k_init<<<dim3((NROW + 255) / 256), b256, 0, stream>>>(minbuf, lossA, NROW);
    k_nchw_to_cl<<<dim3((RECON + 255) / 256), b256, 0, stream>>>(x, x_cl, 3, 256, 256, RECON);

    auto wt = [&](const float* wp, bf16_t* Bp, int Co, int Ci, int kh, int kw, int Kp, int tm) {
        int Cp = ((Co + 127) / 128) * 128;
        k_wtrans<<<dim3((Cp * Kp + 255) / 256), b256, 0, stream>>>(wp, Bp, Co, Ci, kh, kw, Kp, Cp, tm);
    };
    wt(enc_w1, Benc1, Hc, 3,  4, 4, 64,   0);
    wt(enc_w2, Benc2, Hc, Hc, 4, 4, 2048, 0);
    wt(enc_w3, Benc3, Hc, Hc, 3, 3, 1152, 0);
    wt(qconv_w, Bqc,  E,  Hc, 1, 1, 128,  0);
    wt(codebook, Bcb, K,  E,  1, 1, 256,  0);
    wt(pq_w,   Bpq,   Hc, E,  1, 1, 256,  0);
    wt(dec_w1, Bdc1,  Hc, Hc, 3, 3, 1152, 0);
    wt(dt_w2,  Bdt2,  Hc, Hc, 4, 4, 2048, 1);
    wt(dt_w3,  Bdt3,  3,  Hc, 4, 4, 2048, 1);
    k_rownorm<<<dim3((K + 255) / 256), b256, 0, stream>>>(codebook, cnorm, K, E);

    auto gemm = [&](const float* srcp, int H, int W, int Cin, int cs,
                    int kh, int kw, int st, int pd, int tm, int Ho, int Wo,
                    bf16_t* Bp, int Kp, int Co, int ds,
                    const float* bp, float al, int rl, int md,
                    float* dp, bf16_t* dbp, unsigned long long* mb) {
        int M = NI * Ho * Wo;
        dim3 grid(M / BM, (Co + 127) / 128);
        k_gemm<<<grid, b256, 0, stream>>>(srcp, H, W, Cin, cs, kh, kw, st, pd, tm,
                                          Ho, Wo, Bp, Kp, Co, ds, bp, al, rl, md, dp, dbp, mb);
    };
    auto gemm_lin = [&](const bf16_t* Ap, bf16_t* Bp, int M, int Kp, int Co, int ds,
                        const float* bp, float al, int rl, int md,
                        float* dp, bf16_t* dbp, unsigned long long* mb) {
        dim3 grid(M / BM, (Co + 127) / 128);
        k_gemm_lin<<<grid, b256, 0, stream>>>(Ap, Bp, Kp, Co, ds, bp, al, rl, md, dp, dbp, mb);
    };

    // ---- encoder (conv layers: gathered implicit GEMM) ----
    gemm(x_cl, 256, 256, 3,  -1, 4, 4, 2, 1, 0, 128, 128, Benc1, 64,   Hc, Hc, enc_b1, 1.f, 1, 0, a1, nullptr, nullptr);
    gemm(a1,   128, 128, Hc,  7, 4, 4, 2, 1, 0, 64,  64,  Benc2, 2048, Hc, Hc, enc_b2, 1.f, 1, 0, a2, nullptr, nullptr);
    gemm(a2,   64,  64,  Hc,  7, 3, 3, 1, 1, 0, 64,  64,  Benc3, 1152, Hc, Hc, enc_b3, 1.f, 0, 0, a3, a3b,    nullptr);

    // ---- 1x1 layers: async double-buffered bf16 A path ----
    gemm_lin(a3b, Bqc, NROW, 128, E, E, qconv_b, 1.f, 0, 0, h, hb, nullptr);

    // VQ distance GEMM (dominant, 68.7 GFLOP) with fused argmin epilogue
    gemm_lin(hb, Bcb, NROW, 256, K, K, cnorm, -2.f, 0, 1, nullptr, nullptr, minbuf);
    k_vq_finish<<<dim3(NROW), b256, 0, stream>>>(minbuf, codebook, h, quant, qtb,
                                                 out + RECON + 1, lossA, E);
    k_loss_final<<<dim3(1), dim3(1), 0, stream>>>(lossA, out + RECON, 1.f / ((float)NROW * (float)E));

    gemm_lin(qtb, Bpq, NROW, 256, Hc, Hc, pq_b, 1.f, 0, 0, qb, nullptr, nullptr);

    // ---- decoder (conv / convT layers) ----
    gemm(qb,    64, 64,  Hc,  7, 3, 3, 1, 1, 0, 64,  64,  Bdc1,  1152, Hc, Hc, dec_b1, 1.f, 1, 0, d1, nullptr, nullptr);
    gemm(d1,    64, 64,  Hc,  7, 4, 4, 1, 1, 1, 128, 128, Bdt2,  2048, Hc, Hc, dt_b2,  1.f, 1, 0, d2, nullptr, nullptr);
    gemm(d2,    128,128, Hc,  7, 4, 4, 1, 1, 1, 256, 256, Bdt3,  2048, 3,  3,  dt_b3,  1.f, 0, 0, recon, nullptr, nullptr);

    k_cl_to_nchw<<<dim3((RECON + 255) / 256), b256, 0, stream>>>(recon, out, 3, 256, 256, RECON);
}
